// HEAD_51127290691996
// MI455X (gfx1250) — matrix operand order repaired and checked by reading, not compiled, not run on hardware
//
#include <hip/hip_runtime.h>

// Single-head causal attention for MI455X (gfx1250, wave32, WMMA).
//   B=512, T=256, C=384, HS=64.  fp32 in/out, bf16 WMMA internally.
//
// d_ws layout (bytes):
//   [0,        16MB)  q_bf16  [B][T][HS]   (pre-scaled by 1/sqrt(HS))
//   [16MB,     32MB)  k_bf16  [B][T][HS]
//   [32MB,     48MB)  vT_bf16 [B][HS][T]
//   [48MB, +144KB )   Wq/Wk/Wv repacked into WMMA B-matrix lane layout

typedef __attribute__((ext_vector_type(16))) __bf16 v16bf;
typedef __attribute__((ext_vector_type(8)))  float  v8f;

#define B_  512
#define T_  256
#define C_  384
#define HS_ 64
#define WPER 24576   // 12 ksteps * 4 ntiles * 32 lanes * 16 elems

union AB16 { v16bf v; __bf16 h[16]; unsigned short u[16]; unsigned long long d[4]; };

// Native convert -> compiler can emit packed v_cvt bf16 ops instead of
// integer bit-twiddling (v_bfe/v_add3 chains seen in the previous round).
__device__ __forceinline__ __bf16 f2bf(float f) { return (__bf16)f; }
__device__ __forceinline__ unsigned short f2bfu(float f) {
  return __builtin_bit_cast(unsigned short, (__bf16)f);
}

// ---------------------------------------------------------------------------
// Repack W [C][HS] fp32 -> bf16 in per-lane WMMA B-matrix layout:
//   region m, index ((kstep*4 + ntile)*32 + lane)*16 + e
//   holds W[kstep*32 + (lane>>4)*8 + (e&7) + (e>>3)*16][ntile*16 + (lane&15)]
// ---------------------------------------------------------------------------
__global__ void wprep_kernel(const float* __restrict__ Wq,
                             const float* __restrict__ Wk,
                             const float* __restrict__ Wv,
                             unsigned short* __restrict__ wpack) {
  int idx = blockIdx.x * blockDim.x + threadIdx.x;
  if (idx >= 3 * WPER) return;
  int m = idx / WPER, rem = idx % WPER;
  int e     = rem & 15;
  int lane  = (rem >> 4) & 31;
  int ntile = (rem >> 9) & 3;
  int kstep = rem >> 11;
  int k = kstep * 32 + (lane >> 4) * 8 + (e & 7) + (e >> 3) * 16;
  int n = ntile * 16 + (lane & 15);
  const float* W = (m == 0) ? Wq : (m == 1) ? Wk : Wv;
  wpack[idx] = f2bfu(W[k * HS_ + n]);
}

// ---------------------------------------------------------------------------
// QKV projection: one wave per 16-row tile of x (M = B*T rows).
// 12 K-steps of 32; A tile (x, cvt to bf16) reused for 12 WMMAs per step.
// ---------------------------------------------------------------------------
__global__ void __launch_bounds__(256) qkv_kernel(
    const float* __restrict__ x, const unsigned short* __restrict__ wpack,
    unsigned short* __restrict__ qb, unsigned short* __restrict__ kb,
    unsigned short* __restrict__ vtb) {
  const int lane  = threadIdx.x & 31;
  const int wave  = threadIdx.x >> 5;
  const int g0    = (blockIdx.x * 8 + wave) * 16;   // first global row of tile
  const int khalf = lane >> 4, nloc = lane & 15;
  const int koff  = khalf * 8;

  const v8f vz = {0.f,0.f,0.f,0.f,0.f,0.f,0.f,0.f};
  v8f accq[4], acck[4], accv[4];
#pragma unroll
  for (int i = 0; i < 4; ++i) { accq[i] = vz; acck[i] = vz; accv[i] = vz; }

  const int arow = g0 + nloc;                       // A-matrix row of this lane

  for (int ks = 0; ks < 12; ++ks) {
    // A: x[arow][ks*32 + koff + 0..7]  and  x[arow][ks*32 + 16 + koff + 0..7]
    const float* base = x + arow * C_ + ks * 32;
    v8f xa = *(const v8f*)(base + koff);
    v8f xb = *(const v8f*)(base + 16 + koff);
    if (ks + 2 < 12) __builtin_prefetch(base + 64, 0, 1);   // global_prefetch_b8
    AB16 a;
#pragma unroll
    for (int e = 0; e < 8; ++e) { a.h[e] = f2bf(xa[e]); a.h[8 + e] = f2bf(xb[e]); }

#pragma unroll
    for (int nt = 0; nt < 4; ++nt) {
      const int o = ((ks * 4 + nt) * 32 + lane) * 16;
      v16bf bq = *(const v16bf*)(wpack + o);
      v16bf bk = *(const v16bf*)(wpack + WPER + o);
      v16bf bv = *(const v16bf*)(wpack + 2 * WPER + o);
      accq[nt] = __builtin_amdgcn_wmma_f32_16x16x32_bf16(false, a.v, false, bq,
                                                         (short)0, accq[nt], false, false);
      acck[nt] = __builtin_amdgcn_wmma_f32_16x16x32_bf16(false, a.v, false, bk,
                                                         (short)0, acck[nt], false, false);
      accv[nt] = __builtin_amdgcn_wmma_f32_16x16x32_bf16(false, a.v, false, bv,
                                                         (short)0, accv[nt], false, false);
    }
  }

  // D layout: elem r -> (M = khalf*8 + r, N = nloc)
  const int Mb   = khalf * 8;
  const int bIdx = g0 / T_;
  const int t0   = (g0 % T_) + Mb;
#pragma unroll
  for (int nt = 0; nt < 4; ++nt) {
    const int n = nt * 16 + nloc;
#pragma unroll
    for (int r = 0; r < 8; ++r) {
      int grow = g0 + Mb + r;
      qb[grow * HS_ + n] = f2bfu(accq[nt][r] * 0.125f);   // fold 1/sqrt(64)
      kb[grow * HS_ + n] = f2bfu(acck[nt][r]);
    }
    // v stored transposed [B][HS][T]: 8 consecutive t -> one packed 16B store
    unsigned short vs[8];
#pragma unroll
    for (int r = 0; r < 8; ++r) vs[r] = f2bfu(accv[nt][r]);
    unsigned long long lo = 0, hi = 0;
#pragma unroll
    for (int r = 0; r < 4; ++r) {
      lo |= (unsigned long long)vs[r]     << (16 * r);
      hi |= (unsigned long long)vs[4 + r] << (16 * r);
    }
    unsigned short* dst = vtb + ((size_t)bIdx * HS_ + n) * T_ + t0;
    *(unsigned long long*)(dst)     = lo;
    *(unsigned long long*)(dst + 4) = hi;
  }
}

// ---------------------------------------------------------------------------
// Flash attention: block = 4 waves x 16 query rows; stream 32-wide key tiles
// with causal masking + online softmax; P routed D->A layout via per-wave LDS.
// ---------------------------------------------------------------------------
__global__ void __launch_bounds__(128) attn_kernel(
    const unsigned short* __restrict__ qb, const unsigned short* __restrict__ kb,
    const unsigned short* __restrict__ vtb, float* __restrict__ out) {
  __shared__ float pbuf[4][16][32];                 // per-wave P staging

  const int lane  = threadIdx.x & 31;
  const int wave  = threadIdx.x >> 5;
  const int b     = blockIdx.x >> 2;
  const int qbase = (blockIdx.x & 3) * 64 + wave * 16;
  const int khalf = lane >> 4, nloc = lane & 15;
  const int koff  = khalf * 16;
  const int Mb    = khalf * 8;

  // Q rows as WMMA A operands (d 0..31 and 32..63)
  AB16 qA0, qA1;
  {
    const unsigned short* qrow = qb + ((size_t)b * T_ + qbase + nloc) * HS_;
    const unsigned long long* p;
    p = (const unsigned long long*)(qrow + koff);      qA0.d[0] = p[0]; qA0.d[1] = p[1];
    p = (const unsigned long long*)(qrow + koff + 8);  qA0.d[2] = p[0]; qA0.d[3] = p[1];
    p = (const unsigned long long*)(qrow + 32 + koff); qA1.d[0] = p[0]; qA1.d[1] = p[1];
    p = (const unsigned long long*)(qrow + 40 + koff); qA1.d[2] = p[0]; qA1.d[3] = p[1];
  }

  const v8f vz = {0.f,0.f,0.f,0.f,0.f,0.f,0.f,0.f};
  float m[8], l[8];
  v8f o[4];
#pragma unroll
  for (int r = 0; r < 8; ++r) { m[r] = -3.0e38f; l[r] = 0.f; }
#pragma unroll
  for (int i = 0; i < 4; ++i) o[i] = vz;

  for (int s0 = 0; s0 <= qbase + 15; s0 += 32) {
    // ---- scores: S[16x32] = Q[16x64] @ K^T ----
    v8f sc[2]; sc[0] = vz; sc[1] = vz;
#pragma unroll
    for (int c = 0; c < 2; ++c) {
      const int srow = s0 + c * 16 + nloc;           // B-matrix column N = key idx
      const unsigned short* krow = kb + ((size_t)b * T_ + srow) * HS_ + khalf * 16;
      v16bf kB0 = *(const v16bf*)(krow);
      v16bf kB1 = *(const v16bf*)(krow + 32);
      sc[c] = __builtin_amdgcn_wmma_f32_16x16x32_bf16(false, qA0.v, false, kB0,
                                                      (short)0, sc[c], false, false);
      sc[c] = __builtin_amdgcn_wmma_f32_16x16x32_bf16(false, qA1.v, false, kB1,
                                                      (short)0, sc[c], false, false);
    }

    // ---- causal mask + online softmax (row = Mb+r lives in one 16-lane half) ----
#pragma unroll
    for (int r = 0; r < 8; ++r) {
      const int t = qbase + Mb + r;
      float s0v = sc[0][r], s1v = sc[1][r];
      if (s0 + nloc      > t) s0v = -3.0e38f;
      if (s0 + 16 + nloc > t) s1v = -3.0e38f;
      float tmax = fmaxf(s0v, s1v);
#pragma unroll
      for (int sh = 8; sh >= 1; sh >>= 1) tmax = fmaxf(tmax, __shfl_xor(tmax, sh, 32));
      const float mn    = fmaxf(m[r], tmax);
      const float alpha = __expf(m[r] - mn);
      const float p0 = __expf(s0v - mn);
      const float p1 = __expf(s1v - mn);
      float rs = p0 + p1;
#pragma unroll
      for (int sh = 8; sh >= 1; sh >>= 1) rs += __shfl_xor(rs, sh, 32);
      l[r] = l[r] * alpha + rs;
      m[r] = mn;
#pragma unroll
      for (int dt = 0; dt < 4; ++dt) o[dt][r] *= alpha;
      pbuf[wave][Mb + r][nloc]      = p0;
      pbuf[wave][Mb + r][16 + nloc] = p1;
    }

    // Same-wave LDS ops are processed in order; just stop compiler reordering.
    asm volatile("" ::: "memory");

    // ---- P (D layout in LDS) -> A layout bf16 ----
    AB16 pA;
#pragma unroll
    for (int e = 0; e < 8; ++e) {
      pA.h[e]     = f2bf(pbuf[wave][nloc][koff + e]);
      pA.h[8 + e] = f2bf(pbuf[wave][nloc][koff + 8 + e]);
    }
    asm volatile("" ::: "memory");

    // ---- O += P @ V  (V^T stored [HS][T] -> contiguous 32B B-operand loads) ----
#pragma unroll
    for (int dt = 0; dt < 4; ++dt) {
      const unsigned short* vcol =
          vtb + ((size_t)b * HS_ + dt * 16 + nloc) * T_ + s0 + khalf * 16;
      v16bf vB = *(const v16bf*)vcol;
      o[dt] = __builtin_amdgcn_wmma_f32_16x16x32_bf16(false, pA.v, false, vB,
                                                      (short)0, o[dt], false, false);
    }
  }

  // ---- normalize + store fp32 (one reciprocal per row, not 32 divides) ----
  float inv[8];
#pragma unroll
  for (int r = 0; r < 8; ++r) inv[r] = __frcp_rn(l[r]);
#pragma unroll
  for (int dt = 0; dt < 4; ++dt) {
#pragma unroll
    for (int r = 0; r < 8; ++r) {
      const int t = qbase + Mb + r;
      const int d = dt * 16 + nloc;
      out[((size_t)b * T_ + t) * HS_ + d] = o[dt][r] * inv[r];
    }
  }
}

// ---------------------------------------------------------------------------
extern "C" void kernel_launch(void* const* d_in, const int* in_sizes, int n_in,
                              void* d_out, int out_size, void* d_ws, size_t ws_size,
                              hipStream_t stream) {
  const float* x  = (const float*)d_in[0];
  const float* Wk = (const float*)d_in[1];
  const float* Wq = (const float*)d_in[2];
  const float* Wv = (const float*)d_in[3];
  float* out = (float*)d_out;

  const size_t qkvElems = (size_t)B_ * T_ * HS_;    // 8,388,608 bf16 each
  unsigned short* qb  = (unsigned short*)d_ws;
  unsigned short* kb  = qb  + qkvElems;
  unsigned short* vtb = kb  + qkvElems;
  unsigned short* wpk = vtb + qkvElems;             // 3*WPER bf16 (~144KB)

  wprep_kernel<<<(3 * WPER + 255) / 256, 256, 0, stream>>>(Wq, Wk, Wv, wpk);
  qkv_kernel<<<(B_ * T_ / 16) / 8, 256, 0, stream>>>(x, wpk, qb, kb, vtb);
  attn_kernel<<<B_ * (T_ / 64), 128, 0, stream>>>(qb, kb, vtb, out);
}
